// SynthesisLayer_5085241278495
// MI455X (gfx1250) — matrix operand order repaired and checked by reading, not compiled, not run on hardware
//
#include <hip/hip_runtime.h>
#include <math.h>

// ---------------------------------------------------------------------------
// SynthesisLayer (StyleGAN3-style) for MI455X / gfx1250, wave32 + WMMA +
// async global->LDS DMA (ASYNCcnt) double-buffered staging.
//   B=8, CIN=COUT=512, H=64, conv 3x3 pad2 -> 66x66, up x2 (12 tap),
//   lrelu*sqrt2 clamp 256, down x2 (12 tap) -> 64x64.
// ---------------------------------------------------------------------------

typedef __attribute__((ext_vector_type(16))) _Float16 v16h;
typedef __attribute__((ext_vector_type(8)))  _Float16 v8h;
typedef __attribute__((ext_vector_type(8)))  float    v8f;

#define NB    8
#define CIN   512
#define COUT  512
#define HH    64
#define YS    66            // conv output spatial (64 + 2*2 - 2)
#define XP    68            // padded x spatial (64 + 2 + 2)
#define KW9   9
#define KROW  4608          // CIN*9
#define SLAB  (3 * 66 * 32) // halves per staged x chunk

// workspace layout (bytes)
#define OFF_STYLES   ((size_t)0)                        // 8*512 f32
#define OFF_SSUM     ((size_t)16384)                    // 8 f32
#define OFF_WNORM    ((size_t)16640)                    // 512 f32
#define OFF_WK       ((size_t)18688)                    // 8*9*512*512 f16 = 37,748,736
#define OFF_XT       (OFF_WK + (size_t)NB*KW9*COUT*CIN*2)        // 8*68*68*512 f16
#define OFF_Y1       (OFF_XT + (size_t)NB*XP*XP*CIN*2)           // 8*512*66*66 f32

// ---------------------------------------------------------------------------
// 1) styles[b,i] = (w[b,:]·aff_w[i,:])/sqrt(512) + aff_b[i];  ssum8[b]=Σ styles²
// ---------------------------------------------------------------------------
__global__ __launch_bounds__(512) void styles_kernel(
    const float* __restrict__ w, const float* __restrict__ aff_w,
    const float* __restrict__ aff_b, float* __restrict__ styles,
    float* __restrict__ ssum8) {
  __shared__ float red[512];
  int b = blockIdx.x, i = threadIdx.x;
  const float* wr = w + b * 512;
  const float* ar = aff_w + (size_t)i * 512;
  float acc = 0.f;
  for (int k = 0; k < 512; ++k) acc += wr[k] * ar[k];
  float st = acc * 0.04419417382415922f /* 1/sqrt(512) */ + aff_b[i];
  styles[b * 512 + i] = st;
  red[i] = st * st;
  __syncthreads();
  for (int off = 256; off > 0; off >>= 1) {
    if (i < off) red[i] += red[i + off];
    __syncthreads();
  }
  if (i == 0) ssum8[b] = red[0];
}

// ---------------------------------------------------------------------------
// 2) wnorm[o] = rsqrt(mean(weight[o,:,:,:]^2))
// ---------------------------------------------------------------------------
__global__ __launch_bounds__(256) void wnorm_kernel(
    const float* __restrict__ weight, float* __restrict__ wnorm) {
  __shared__ float red[256];
  int o = blockIdx.x, tid = threadIdx.x;
  const float* wrow = weight + (size_t)o * KROW;
  float s = 0.f;
  for (int e = tid; e < KROW; e += 256) { float v = wrow[e]; s += v * v; }
  red[tid] = s;
  __syncthreads();
  for (int off = 128; off > 0; off >>= 1) {
    if (tid < off) red[tid] += red[tid + off];
    __syncthreads();
  }
  if (tid == 0) wnorm[o] = rsqrtf(red[0] * (1.0f / (float)KROW));
}

// ---------------------------------------------------------------------------
// 3) wk -> f16, layout [b][t][o][i] (i contiguous: A-fragment = 2x16B loads)
// ---------------------------------------------------------------------------
__global__ __launch_bounds__(256) void wk_kernel(
    const float* __restrict__ weight, const float* __restrict__ wnorm,
    const float* __restrict__ styles, const float* __restrict__ ssum8,
    _Float16* __restrict__ wkb) {
  __shared__ float red[256];
  __shared__ float sdco;
  int o = blockIdx.x, b = blockIdx.y, tid = threadIdx.x;
  float ssum = 0.f;
#pragma unroll
  for (int k = 0; k < NB; ++k) ssum += ssum8[k];
  float sinv = rsqrtf(ssum * (1.0f / (float)(NB * CIN)));
  float wno = wnorm[o];
  const float* wrow = weight + (size_t)o * KROW;
  const float* srow = styles + b * 512;

  float sq = 0.f;
  for (int e = tid; e < KROW; e += 256) {
    int iv = e / 9;
    float val = wrow[e] * wno * (srow[iv] * sinv);
    sq += val * val;
  }
  red[tid] = sq;
  __syncthreads();
  for (int off = 128; off > 0; off >>= 1) {
    if (tid < off) red[tid] += red[tid + off];
    __syncthreads();
  }
  if (tid == 0) sdco = rsqrtf(red[0] + 1e-8f);
  __syncthreads();
  float dco = sdco;
  for (int e = tid; e < KROW; e += 256) {
    int iv = e / 9, t = e - iv * 9;
    float val = wrow[e] * wno * (srow[iv] * sinv) * dco;
    wkb[(((size_t)b * KW9 + t) * COUT + o) * CIN + iv] = (_Float16)val;
  }
}

// ---------------------------------------------------------------------------
// 3b) x -> f16, transposed + zero-padded:  xT[b][pr 0..67][pc 0..67][i]
//     so conv staging is raw contiguous bytes (async-DMA friendly).
// ---------------------------------------------------------------------------
__global__ __launch_bounds__(256) void xprep_kernel(
    const float* __restrict__ x, _Float16* __restrict__ xT) {
  int pos = blockIdx.x;                  // 0..68*68-1
  int b = blockIdx.y;
  int pr = pos / XP, pc = pos - XP * pr;
  int gr = pr - 2, gc = pc - 2;
  _Float16* dst = xT + (((size_t)b * XP + pr) * XP + pc) * CIN;
  bool inb = ((unsigned)gr < (unsigned)HH) && ((unsigned)gc < (unsigned)HH);
  for (int i = threadIdx.x; i < CIN; i += 256) {
    float v = inb ? x[(((size_t)b * CIN + i) * HH + gr) * HH + gc] : 0.f;
    dst[i] = (_Float16)v;
  }
}

// ---------------------------------------------------------------------------
// 4) implicit-GEMM conv via v_wmma_f32_16x16x32_f16 with double-buffered
//    GLOBAL_LOAD_ASYNC_TO_LDS_B128 staging of the x slab.
//    Block: (b, 64-ch tile, row r, 64-col tile). 8 waves = 4(M) x 2(N),
//    each wave: 2 f32 16x16 accumulators, 18 WMMA per 32-cin chunk.
// ---------------------------------------------------------------------------
__global__ __launch_bounds__(256) void conv_wmma_kernel(
    const _Float16* __restrict__ xT, const _Float16* __restrict__ wkb,
    float* __restrict__ y1) {
  // staged slab: [row(3)][col(66)][i(32)] f16 per buffer; i contiguous so a
  // B-fragment lane (n=lane&15, K=h*16+j) is one aligned 32B run.
  __shared__ __align__(32) _Float16 xs[2][SLAB];

  int tid  = threadIdx.x;
  int lane = tid & 31;
  int wv   = tid >> 5;
  int wm   = wv & 3;
  int wn   = wv >> 2;
  int r    = blockIdx.x >> 1;
  int ct   = blockIdx.x & 1;
  int o0   = blockIdx.y * 64;
  int b    = blockIdx.z;
  int c0   = ct * 64;
  int m    = lane & 15;
  int h    = lane >> 4;

  // slab corner in padded coords: (row r, col c0) -> covers rows r..r+2,
  // cols c0..c0+65 (ct=1 over-reads stay inside the workspace; those lanes'
  // stores are guarded off below).
  const _Float16* gslab = xT + (((size_t)b * XP + r) * XP + c0) * CIN;
  uint32_t lds0 = (uint32_t)(uintptr_t)(void*)&xs[0][0];

  // ---- prologue: async-issue chunk 0 into buffer 0
  {
    uint64_t sa = (uint64_t)(uintptr_t)gslab;
    for (int e = tid; e < 792; e += 256) {        // 198 runs x 4 xB128
      int run = e >> 2, j = e & 3;
      int rr = run / 66, cc = run - 66 * rr;
      uint32_t ldsa = lds0 + (uint32_t)((((rr * 66 + cc) * 32) + j * 8) * 2);
      uint32_t voff = (uint32_t)((((rr * XP + cc) * CIN) + j * 8) * 2);
      asm volatile("global_load_async_to_lds_b128 %0, %1, %2"
                   :: "v"(ldsa), "v"(voff), "s"(sa) : "memory");
    }
  }

  v8f acc0 = {};
  v8f acc1 = {};

  for (int ic = 0; ic < CIN / 32; ++ic) {
    asm volatile("s_wait_asynccnt 0x0" ::: "memory");
    __syncthreads();   // all waves' DMA done -> slab(ic&1) readable by all

    // ---- async-issue chunk ic+1 into the other buffer (overlaps WMMAs)
    if (ic + 1 < CIN / 32) {
      uint64_t sa = (uint64_t)(uintptr_t)(gslab + (ic + 1) * 32);
      uint32_t ldsb = lds0 + (uint32_t)(((ic + 1) & 1) * (SLAB * 2));
      for (int e = tid; e < 792; e += 256) {
        int run = e >> 2, j = e & 3;
        int rr = run / 66, cc = run - 66 * rr;
        uint32_t ldsa = ldsb + (uint32_t)((((rr * 66 + cc) * 32) + j * 8) * 2);
        uint32_t voff = (uint32_t)((((rr * XP + cc) * CIN) + j * 8) * 2);
        asm volatile("global_load_async_to_lds_b128 %0, %1, %2"
                     :: "v"(ldsa), "v"(voff), "s"(sa) : "memory");
      }
    }

    const _Float16* xcur = xs[ic & 1];
#pragma unroll
    for (int t = 0; t < KW9; ++t) {
      int dy = t / 3, dx = t - 3 * (t / 3);

      // A fragment (contiguous k order, same as B: lanes 0-15 K=0-15; 16-31 K=16-31)
      const _Float16* ap =
          wkb + (((size_t)b * KW9 + t) * COUT + (o0 + wm * 16 + m)) * CIN + ic * 32;
      v8h alo = *(const v8h*)(ap + h * 16);
      v8h ahi = *(const v8h*)(ap + h * 16 + 8);
      v16h a;
#pragma unroll
      for (int j = 0; j < 8; ++j) { a[j] = alo[j]; a[8 + j] = ahi[j]; }

      // B fragments: staged col = (c - c0) + dx  (padding already materialized)
      int colA = wn * 32 + 0  + m + dx;
      int colB = wn * 32 + 16 + m + dx;
      v16h b0 = *(const v16h*)&xcur[(dy * 66 + colA) * 32 + h * 16];
      v16h b1 = *(const v16h*)&xcur[(dy * 66 + colB) * 32 + h * 16];

      acc0 = __builtin_amdgcn_wmma_f32_16x16x32_f16(
          false, a, false, b0, (short)0, acc0, false, false);
      acc1 = __builtin_amdgcn_wmma_f32_16x16x32_f16(
          false, a, false, b1, (short)0, acc1, false, false);
    }
    // no trailing barrier: next iteration's wait+barrier covers the WAR.
  }

  // ---- store: C/D layout -> VGPR v holds M = v + 8*h, N = lane&15
  int cA = c0 + wn * 32 + m;
  int cB = cA + 16;
#pragma unroll
  for (int v = 0; v < 8; ++v) {
    int oc = o0 + wm * 16 + v + h * 8;
    if (cA < YS) y1[(((size_t)b * COUT + oc) * YS + r) * YS + cA] = acc0[v];
    if (cB < YS) y1[(((size_t)b * COUT + oc) * YS + r) * YS + cB] = acc1[v];
  }
}

// ---------------------------------------------------------------------------
// 5) fused filtered_lrelu per (b,c) plane (12-row z ring in LDS).
// ---------------------------------------------------------------------------
__global__ __launch_bounds__(256) void flrelu_kernel(
    const float* __restrict__ y1, const float* __restrict__ bias,
    const float* __restrict__ fu, const float* __restrict__ fd,
    float* __restrict__ out) {
  __shared__ float ybuf[YS * YS];
  __shared__ float zring[12][144];
  __shared__ float vrow[YS];
  __shared__ float w1row[138];
  __shared__ float sfu[12], sfd[12];

  int tid = threadIdx.x;
  int plane = blockIdx.x;                // b*512 + c
  int c = plane & (COUT - 1);

  if (tid < 12) { sfu[tid] = fu[tid] * 2.0f; sfd[tid] = fd[tid]; }
  float bc = bias[c];
  const float* src = y1 + (size_t)plane * (YS * YS);
  for (int e = tid; e < YS * YS; e += 256) ybuf[e] = src[e] + bc;
  __syncthreads();

  int next_ry = 0;
  for (int oy = 0; oy < 64; ++oy) {
    int need = 2 * oy + 12;
    while (next_ry < need) {
      int ry = next_ry;
      for (int xcol = tid; xcol < YS; xcol += 256) {
        float acc = 0.f;
#pragma unroll
        for (int s = 0; s < 12; ++s) {
          int u = ry + s - 9;
          if ((u & 1) == 0 && u >= 0 && u <= 130)
            acc += sfu[s] * ybuf[(u >> 1) * YS + xcol];
        }
        vrow[xcol] = acc;
      }
      __syncthreads();
      for (int rx = tid; rx < 138; rx += 256) {
        float acc = 0.f;
#pragma unroll
        for (int s = 0; s < 12; ++s) {
          int u = rx + s - 9;
          if ((u & 1) == 0 && u >= 0 && u <= 130)
            acc += sfu[s] * vrow[u >> 1];
        }
        float z = (acc >= 0.f) ? acc : 0.2f * acc;
        z *= 1.4142135623730951f;
        z = fminf(fmaxf(z, -256.f), 256.f);
        zring[ry % 12][rx] = z;
      }
      __syncthreads();
      ++next_ry;
    }
    for (int rx = tid; rx < 138; rx += 256) {
      float acc = 0.f;
#pragma unroll
      for (int t = 0; t < 12; ++t)
        acc += sfd[t] * zring[(2 * oy + t) % 12][rx];
      w1row[rx] = acc;
    }
    __syncthreads();
    for (int ox = tid; ox < 64; ox += 256) {
      float acc = 0.f;
#pragma unroll
      for (int t = 0; t < 12; ++t)
        acc += sfd[t] * w1row[2 * ox + t];
      out[(size_t)plane * 4096 + oy * 64 + ox] = acc;
    }
    __syncthreads();
  }
}

// ---------------------------------------------------------------------------
extern "C" void kernel_launch(void* const* d_in, const int* in_sizes, int n_in,
                              void* d_out, int out_size, void* d_ws, size_t ws_size,
                              hipStream_t stream) {
  const float* x      = (const float*)d_in[0];
  const float* w      = (const float*)d_in[1];
  const float* weight = (const float*)d_in[2];
  const float* aff_w  = (const float*)d_in[3];
  const float* aff_b  = (const float*)d_in[4];
  const float* bias   = (const float*)d_in[5];
  const float* fu     = (const float*)d_in[6];
  const float* fd     = (const float*)d_in[7];
  float* out = (float*)d_out;

  char* ws = (char*)d_ws;
  float*     styles = (float*)(ws + OFF_STYLES);
  float*     ssum8  = (float*)(ws + OFF_SSUM);
  float*     wnorm  = (float*)(ws + OFF_WNORM);
  _Float16*  wkb    = (_Float16*)(ws + OFF_WK);
  _Float16*  xT     = (_Float16*)(ws + OFF_XT);
  float*     y1     = (float*)(ws + OFF_Y1);

  styles_kernel<<<NB, 512, 0, stream>>>(w, aff_w, aff_b, styles, ssum8);
  wnorm_kernel<<<COUT, 256, 0, stream>>>(weight, wnorm);
  wk_kernel<<<dim3(COUT, NB), 256, 0, stream>>>(weight, wnorm, styles, ssum8, wkb);
  xprep_kernel<<<dim3(XP * XP, NB), 256, 0, stream>>>(x, xT);
  conv_wmma_kernel<<<dim3(YS * 2, COUT / 64, NB), 256, 0, stream>>>(xT, wkb, y1);
  flrelu_kernel<<<NB * COUT, 256, 0, stream>>>(y1, bias, fu, fd, out);
}
